// RelationalLayerModel_29875792511275
// MI455X (gfx1250) — compile-verified, matrix operand order checked by reading, not run on hardware
//
#include <hip/hip_runtime.h>

// RelationalLayerModel: MB=64, C=24, D=8 (s2=64), QS=128, OUT=28, H=256, IN=180
//
// Strategy (compute-bound, ~103 GFLOP after rank-splitting layer 1):
//   1) conv_w:      Wg2/3/4 f32 -> f16 (WMMA B operands)
//   2) precompute:  A1[b,q,:] = Wg1[:, :26] @ x_flat[b,q];  B1[b,p,:] = Wg1[:, 26:] @ x_j[b,p]
//   3) g_main:      per (b, 32-row pair tile): h1 = relu(A1+B1+bg1) into LDS (f16),
//                   then 3x fused [32x256]x[256x256] GEMM+ReLU via v_wmma_f32_16x16x32_f16,
//                   last layer fuses relu + row-sum -> partial[b,tile,:]
//   4) f_mlp:       sum partials, 2x 256x256 + 256x28 MLP, log_softmax
//
// ws layout (bytes):  A1 @0 (4MB) | B1 @4MB (4MB) | Wh @8MB (384KB) | part @8.5MB (8MB)

typedef __attribute__((ext_vector_type(16))) _Float16 v16h;
typedef __attribute__((ext_vector_type(8)))  _Float16 v8h;
typedef __attribute__((ext_vector_type(8)))  float    v8f;

#define HDIM   256
#define STRIDE 264   // 256 + 8 f16 pad (16B aligned rows, bank skew)

// ---------------------------------------------------------------- weights->f16
__global__ __launch_bounds__(256) void conv_w_kernel(
    const float* __restrict__ w2, const float* __restrict__ w3,
    const float* __restrict__ w4, _Float16* __restrict__ out)
{
  int i = blockIdx.x * 256 + threadIdx.x;           // 0 .. 3*65536-1
  const float* src = (i < 65536) ? w2 : (i < 131072) ? w3 : w4;
  out[i] = (_Float16)src[i & 65535];
}

// ------------------------------------------------- layer-1 rank-split precompute
// blocks 0..4095:   A1 row (b = blk/64, q = blk%64)
// blocks 4096..8191: B1 row (b, p)
__global__ __launch_bounds__(256) void precompute_ab_kernel(
    const float* __restrict__ x, const float* __restrict__ qst,
    const float* __restrict__ Wg1,
    float* __restrict__ A1, float* __restrict__ B1)
{
  const int blk = blockIdx.x;
  const int t   = threadIdx.x;                      // output feature h
  const bool isB = blk >= 4096;
  const int bi = (blk & 4095) >> 6;
  const int s  = blk & 63;
  const float cx = ((float)(s >> 3) - 4.0f) * 0.25f;
  const float cy = ((float)(s & 7)  - 4.0f) * 0.25f;

  const float* w = Wg1 + t * 180 + (isB ? 26 : 0);
  const float* xb = x + (bi * 24) * 64 + s;
  float acc = 0.0f;
  #pragma unroll
  for (int k = 0; k < 24; ++k) acc += w[k] * xb[k * 64];
  acc += w[24] * cx + w[25] * cy;
  if (isB) {
    const float* q = qst + bi * 128;
    for (int k = 0; k < 128; ++k) acc += w[26 + k] * q[k];
    B1[(bi * 64 + s) * HDIM + t] = acc;
  } else {
    A1[(bi * 64 + s) * HDIM + t] = acc;
  }
}

// ------------------------------------------------------------- fused g-MLP core
// grid: (128, 64)  x = tile (p = tile/2, q-half = tile&1), y = batch
// 256 threads = 8 waves; wave: rb = w&1 (16-row block), ch = w>>1 (64-col chunk)
__global__ __launch_bounds__(256) void g_main_kernel(
    const float* __restrict__ A1, const float* __restrict__ B1,
    const _Float16* __restrict__ Wh,
    const float* __restrict__ bg1, const float* __restrict__ bg2,
    const float* __restrict__ bg3, const float* __restrict__ bg4,
    float* __restrict__ part)
{
  __shared__ __align__(16) _Float16 buf0[32 * STRIDE];
  __shared__ __align__(16) _Float16 buf1[32 * STRIDE];
  __shared__ float red[256 * 4];

  const int tile = blockIdx.x;
  const int b    = blockIdx.y;
  const int p    = tile >> 1;
  const int qh   = tile & 1;
  const int t    = threadIdx.x;

  // ---- layer 1: h1[q][c] = relu(A1[b][q][c] + B1[b][p][c] + bg1[c]) -> f16 LDS
  {
    const float bv = B1[(b * 64 + p) * HDIM + t] + bg1[t];
    const float* ap = A1 + (b * 64 + qh * 32) * HDIM + t;
    #pragma unroll 4
    for (int i = 0; i < 32; ++i) {
      float v = ap[i * HDIM] + bv;
      buf0[i * STRIDE + t] = (_Float16)fmaxf(v, 0.0f);
    }
  }
  __syncthreads();

  const int wave = t >> 5;
  const int lane = t & 31;
  const int rb   = wave & 1;        // which 16-row block
  const int ch   = wave >> 1;       // which 64-col chunk
  const int n    = lane & 15;       // A row (M) / B col (N) for this lane
  const int half = lane >> 4;       // K-half selector per ISA fragment layout
  const int arow = rb * 16 + n;
  const int colbase = ch * 64;

  for (int L = 0; L < 3; ++L) {
    // ping-pong buffer select via ternary (no LDS-pointer arrays: those become
    // illegal addrspacecast static initializers)
    const _Float16* in  = (L & 1) ? buf1 : buf0;
    _Float16*       out = (L & 1) ? buf0 : buf1;
    const _Float16* W   = Wh + L * HDIM * HDIM;
    const float*  bias  = (L == 0) ? bg2 : (L == 1) ? bg3 : bg4;
    (void)bg1;

    v8f acc[4];
    #pragma unroll
    for (int nt = 0; nt < 4; ++nt) {
      float bv = bias[colbase + nt * 16 + n];   // column-only -> same in all 8 rows
      #pragma unroll
      for (int r = 0; r < 8; ++r) acc[nt][r] = bv;
    }

    #pragma unroll
    for (int k = 0; k < 8; ++k) {
      const int k0 = k * 32;
      union { v16h v; v8h h[2]; } aF;
      const _Float16* apnt = in + arow * STRIDE + k0 + half * 8;
      aF.h[0] = *(const v8h*)(apnt);        // VGPR0-3: K = k0 + half*8 + 0..7
      aF.h[1] = *(const v8h*)(apnt + 16);   // VGPR4-7: K = k0 + 16 + half*8 + 0..7
      #pragma unroll
      for (int nt = 0; nt < 4; ++nt) {
        union { v16h v; v8h h[2]; } bF;     // W is [N][K] row-major == B operand
        const _Float16* wp = W + (colbase + nt * 16 + n) * HDIM + k0 + half * 8;
        bF.h[0] = *(const v8h*)(wp);
        bF.h[1] = *(const v8h*)(wp + 16);
        acc[nt] = __builtin_amdgcn_wmma_f32_16x16x32_f16(
            false, aF.v, false, bF.v, (short)0, acc[nt], false, false);
      }
    }

    if (L < 2) {
      // relu -> f16 -> ping-pong buffer
      #pragma unroll
      for (int nt = 0; nt < 4; ++nt) {
        const int coln = colbase + nt * 16 + n;
        #pragma unroll
        for (int r = 0; r < 8; ++r) {
          const int row = rb * 16 + half * 8 + r;   // D layout: M = 8*half + r
          out[row * STRIDE + coln] = (_Float16)fmaxf(acc[nt][r], 0.0f);
        }
      }
      __syncthreads();
    } else {
      // h4: relu + sum over the 32 rows of this tile (deterministic)
      #pragma unroll
      for (int nt = 0; nt < 4; ++nt) {
        float s = 0.0f;
        #pragma unroll
        for (int r = 0; r < 8; ++r) s += fmaxf(acc[nt][r], 0.0f);
        red[(colbase + nt * 16 + n) * 4 + rb * 2 + half] = s;
      }
      __syncthreads();
      float tot = red[t * 4] + red[t * 4 + 1] + red[t * 4 + 2] + red[t * 4 + 3];
      part[(b * 128 + tile) * HDIM + t] = tot;
    }
  }
}

// --------------------------------------------- reduce + f-MLP + log_softmax
__global__ __launch_bounds__(256) void f_mlp_kernel(
    const float* __restrict__ part,
    const float* __restrict__ Wf1, const float* __restrict__ bf1,
    const float* __restrict__ Wf2, const float* __restrict__ bf2,
    const float* __restrict__ Wf3, const float* __restrict__ bf3,
    float* __restrict__ out)
{
  __shared__ float s0[256];
  __shared__ float s1[256];
  __shared__ float lg[32];
  const int b = blockIdx.x, t = threadIdx.x;

  float s = 0.0f;
  const float* pp = part + b * 128 * HDIM + t;
  for (int i = 0; i < 128; ++i) s += pp[i * HDIM];   // x_g[b][t]
  s0[t] = s;
  __syncthreads();

  float a = bf1[t];
  for (int k = 0; k < 256; ++k) a += Wf1[t * 256 + k] * s0[k];
  s1[t] = fmaxf(a, 0.0f);
  __syncthreads();

  a = bf2[t];
  for (int k = 0; k < 256; ++k) a += Wf2[t * 256 + k] * s1[k];
  s0[t] = fmaxf(a, 0.0f);
  __syncthreads();

  if (t < 28) {
    a = bf3[t];
    for (int k = 0; k < 256; ++k) a += Wf3[t * 256 + k] * s0[k];
    lg[t] = a;
  }
  __syncthreads();

  if (t == 0) {
    float m = -1e30f;
    for (int o = 0; o < 28; ++o) m = fmaxf(m, lg[o]);
    float se = 0.0f;
    for (int o = 0; o < 28; ++o) se += expf(lg[o] - m);
    const float lse = m + logf(se);
    for (int o = 0; o < 28; ++o) out[b * 28 + o] = lg[o] - lse;
  }
}

extern "C" void kernel_launch(void* const* d_in, const int* in_sizes, int n_in,
                              void* d_out, int out_size, void* d_ws, size_t ws_size,
                              hipStream_t stream) {
  (void)in_sizes; (void)n_in; (void)out_size; (void)ws_size;
  const float* x   = (const float*)d_in[0];
  const float* qst = (const float*)d_in[1];
  const float* Wg1 = (const float*)d_in[2];
  const float* bg1 = (const float*)d_in[3];
  const float* Wg2 = (const float*)d_in[4];
  const float* bg2 = (const float*)d_in[5];
  const float* Wg3 = (const float*)d_in[6];
  const float* bg3 = (const float*)d_in[7];
  const float* Wg4 = (const float*)d_in[8];
  const float* bg4 = (const float*)d_in[9];
  const float* Wf1 = (const float*)d_in[10];
  const float* bf1 = (const float*)d_in[11];
  const float* Wf2 = (const float*)d_in[12];
  const float* bf2 = (const float*)d_in[13];
  const float* Wf3 = (const float*)d_in[14];
  const float* bf3 = (const float*)d_in[15];
  float* out = (float*)d_out;

  char* ws = (char*)d_ws;
  float*    A1   = (float*)(ws);                                  // 4 MB
  float*    B1   = (float*)(ws + (size_t)4 * 1024 * 1024);        // 4 MB
  _Float16* Wh   = (_Float16*)(ws + (size_t)8 * 1024 * 1024);     // 384 KB
  float*    part = (float*)(ws + (size_t)8 * 1024 * 1024 + 512 * 1024); // 8 MB

  conv_w_kernel<<<768, 256, 0, stream>>>(Wg2, Wg3, Wg4, Wh);
  precompute_ab_kernel<<<8192, 256, 0, stream>>>(x, qst, Wg1, A1, B1);
  g_main_kernel<<<dim3(128, 64), 256, 0, stream>>>(A1, B1, Wh, bg1, bg2, bg3, bg4, part);
  f_mlp_kernel<<<64, 256, 0, stream>>>(part, Wf1, bf1, Wf2, bf2, Wf3, bf3, out);
}
